// distanceNet_8899172237856
// MI455X (gfx1250) — compile-verified, matrix operand order checked by reading, not run on hardware
//
#include <hip/hip_runtime.h>

// CDNA5 / gfx1250, wave32.
typedef __attribute__((ext_vector_type(16))) _Float16 v16h;
typedef __attribute__((ext_vector_type(8)))  float    v8f;
typedef __attribute__((ext_vector_type(4)))  float    v4f;

#define D_FEAT 128
#define WAVES_PER_BLOCK 8
#define EDGES_PER_BLOCK (WAVES_PER_BLOCK * 16)

// ---------------------------------------------------------------------------
// Kernel 0: init per-node segment-max (as u32 bit pattern, e>0 so 0u works)
// and segment-sum accumulators.
// ---------------------------------------------------------------------------
__global__ void init_nodes(unsigned int* __restrict__ mmax,
                           float* __restrict__ ssum, int n) {
    int i = blockIdx.x * blockDim.x + threadIdx.x;
    if (i < n) { mmax[i] = 0u; ssum[i] = 0.0f; }
}

// ---------------------------------------------------------------------------
// Kernel 1: per-edge score e = exp(-0.01 * L1(feats[src], feats[dst])).
// One wave handles 16 edges. Lane l (row = l&15, hi = l>>4) loads 16 floats
// of edge `row`'s features per 32-wide chunk; lanes (m, m+16) jointly cover
// the 32 K-elements of A-matrix row m. B = all-ones f16, so D[m][*] = L1 sum
// for edge m, accumulated in f32 across 4 WMMA ops (K chunks of 32).
// B==ones makes the result invariant to the exact K position of each element,
// so no transpose/swizzle of A is needed.
// ---------------------------------------------------------------------------
__global__ __launch_bounds__(256) void edge_scores_wmma(
    const float* __restrict__ feats,
    const int*   __restrict__ src,
    const int*   __restrict__ dst,
    float*       __restrict__ e_out,
    unsigned int* __restrict__ mmax,
    int n_edges)
{
    __shared__ float red[WAVES_PER_BLOCK][8][32];

    const int lane = threadIdx.x & 31;
    const int wave = threadIdx.x >> 5;
    const int row  = lane & 15;   // edge within the 16-edge tile
    const int hi   = lane >> 4;   // which 16-feature half of the 32-chunk

    const int ebase = (blockIdx.x * WAVES_PER_BLOCK + wave) * 16;
    int eidx = ebase + row;
    if (eidx >= n_edges) eidx = n_edges - 1;   // clamp: keep EXEC all-ones

    const int si = src[eidx];
    const int di = dst[eidx];
    const float* sp = feats + (size_t)si * D_FEAT + hi * 16;
    const float* dp = feats + (size_t)di * D_FEAT + hi * 16;

    v16h ones;
#pragma unroll
    for (int j = 0; j < 16; ++j) ones[j] = (_Float16)1.0f;

    v8f acc = {};
#pragma unroll
    for (int c = 0; c < 4; ++c) {
        const float* spc = sp + c * 32;
        const float* dpc = dp + c * 32;
        v4f s0 = *(const v4f*)(spc + 0);
        v4f s1 = *(const v4f*)(spc + 4);
        v4f s2 = *(const v4f*)(spc + 8);
        v4f s3 = *(const v4f*)(spc + 12);
        v4f d0 = *(const v4f*)(dpc + 0);
        v4f d1 = *(const v4f*)(dpc + 4);
        v4f d2 = *(const v4f*)(dpc + 8);
        v4f d3 = *(const v4f*)(dpc + 12);

        v16h a;
#pragma unroll
        for (int j = 0; j < 4; ++j) {
            a[j +  0] = (_Float16)__builtin_fabsf(s0[j] - d0[j]);
            a[j +  4] = (_Float16)__builtin_fabsf(s1[j] - d1[j]);
            a[j +  8] = (_Float16)__builtin_fabsf(s2[j] - d2[j]);
            a[j + 12] = (_Float16)__builtin_fabsf(s3[j] - d3[j]);
        }
        // (neg_a, A, neg_b, B, c_mod, C, reuse_a, reuse_b)
        acc = __builtin_amdgcn_wmma_f32_16x16x32_f16(
            false, a, false, ones, (short)0, acc, false, false);
    }

    // D layout: component r of lanes 0..15 -> (M=r, N=lane); lanes 16..31 ->
    // (M=r+8). All columns equal (B ones) -> read column 0 of each half.
#pragma unroll
    for (int j = 0; j < 8; ++j) red[wave][j][lane] = acc[j];
    __syncthreads();

    if (lane < 16 && (ebase + lane) < n_edges) {
        const float l1 = red[wave][lane & 7][(lane >> 3) * 16];
        const float ev = expf(-0.01f * l1);
        e_out[ebase + lane] = ev;
        // positive floats order-preserve as u32 bit patterns
        atomicMax(mmax + di, __float_as_uint(ev));
    }
}

// ---------------------------------------------------------------------------
// Kernel 2: ex = exp(e - m[dst]) -> d_out; segment-sum via f32 atomics.
// ---------------------------------------------------------------------------
__global__ void softmax_numer(const float* __restrict__ e,
                              const int*   __restrict__ dst,
                              const unsigned int* __restrict__ mmax,
                              float* __restrict__ ex_out,
                              float* __restrict__ ssum, int n) {
    int i = blockIdx.x * blockDim.x + threadIdx.x;
    if (i >= n) return;
    const int d = dst[i];
    const float m  = __uint_as_float(mmax[d]);
    const float ex = expf(e[i] - m);
    ex_out[i] = ex;
    atomicAdd(ssum + d, ex);
}

// ---------------------------------------------------------------------------
// Kernel 3: normalize.
// ---------------------------------------------------------------------------
__global__ void softmax_norm(float* __restrict__ out,
                             const int* __restrict__ dst,
                             const float* __restrict__ ssum, int n) {
    int i = blockIdx.x * blockDim.x + threadIdx.x;
    if (i >= n) return;
    out[i] = out[i] / ssum[dst[i]];
}

// ---------------------------------------------------------------------------
extern "C" void kernel_launch(void* const* d_in, const int* in_sizes, int n_in,
                              void* d_out, int out_size, void* d_ws, size_t ws_size,
                              hipStream_t stream) {
    const float* feats = (const float*)d_in[0];
    const int*   src   = (const int*)d_in[1];
    const int*   dst   = (const int*)d_in[2];

    const int n_edges = in_sizes[1];
    const int n_nodes = in_sizes[0] / D_FEAT;

    // Workspace layout: e[E] | mmax[N] (u32) | ssum[N]
    float*        e_ws = (float*)d_ws;
    unsigned int* mmax = (unsigned int*)(e_ws + n_edges);
    float*        ssum = (float*)(mmax + n_nodes);

    float* out = (float*)d_out;

    init_nodes<<<(n_nodes + 255) / 256, 256, 0, stream>>>(mmax, ssum, n_nodes);

    const int n_blocks = (n_edges + EDGES_PER_BLOCK - 1) / EDGES_PER_BLOCK;
    edge_scores_wmma<<<n_blocks, 256, 0, stream>>>(feats, src, dst, e_ws, mmax,
                                                   n_edges);

    softmax_numer<<<(n_edges + 255) / 256, 256, 0, stream>>>(e_ws, dst, mmax,
                                                             out, ssum, n_edges);

    softmax_norm<<<(n_edges + 255) / 256, 256, 0, stream>>>(out, dst, ssum,
                                                            n_edges);
}
